// DLRM_Net_3281355014703
// MI455X (gfx1250) — compile-verified, matrix operand order checked by reading, not run on hardware
//
#include <hip/hip_runtime.h>
#include <hip/hip_bf16.h>
#include <math.h>

typedef __attribute__((ext_vector_type(16))) __bf16 v16bf;
typedef __attribute__((ext_vector_type(8)))  float  v8f;
typedef __attribute__((ext_vector_type(4)))  __bf16 v4bf;

#define T_TABLES 26
#define BATCH    8192
#define BAG_L    16
#define EMB_ROWS 100000
#define EMB_D    64
#define ZCOLS    1728   // 64 + 26*64, multiple of 32

union FragBF { v16bf v; uint4 q[2]; };

// ---- CDNA5 async global->LDS copy (ASYNCcnt-tracked, no VGPR round trip) ----
__device__ __forceinline__ void async_copy_b128(unsigned lds_byte_addr, const void* gaddr) {
  // GLOBAL_LOAD_ASYNC_TO_LDS_B128: VDST = LDS byte address VGPR, VADDR = 64-bit global addr
  asm volatile("global_load_async_to_lds_b128 %0, %1, off"
               :: "v"(lds_byte_addr), "v"(gaddr)
               : "memory");
}
__device__ __forceinline__ void wait_async0() {
  asm volatile("s_wait_asynccnt 0" ::: "memory");
}
// generic->LDS byte address: low 32 bits of the generic address
__device__ __forceinline__ unsigned lds_addr32(const void* p) {
  return (unsigned)(unsigned long long)p;
}

// ---------- f32 -> bf16 convert with K padded to Kp (zero fill) ----------
__global__ void pad_convert_bf16(const float* __restrict__ in, __bf16* __restrict__ out,
                                 int rows, int K, int Kp) {
  int i = blockIdx.x * blockDim.x + threadIdx.x;
  if (i >= rows * Kp) return;
  int r = i / Kp, k = i - r * Kp;
  float v = (k < K) ? in[(size_t)r * K + k] : 0.0f;
  out[i] = (__bf16)v;
}

// ---------- embedding bag-sum, writes bf16 into concat z layout ----------
// 16 lanes per bag, each lane owns one float4 chunk of D=64. Pure HBM gather:
// 872 MB of random reads -> float4 per lane, 256B per row across the group.
__global__ void emb_pool_kernel(const float* __restrict__ emb, const int* __restrict__ idx,
                                __bf16* __restrict__ zb) {
  int bag = blockIdx.x * 16 + (threadIdx.x >> 4);   // 0 .. T*B-1 (exact)
  int c   = threadIdx.x & 15;                       // float4 chunk id
  int t   = bag / BATCH;
  int b   = bag - t * BATCH;
  const int*   ip   = idx + (size_t)t * (BATCH * BAG_L) + (size_t)b * BAG_L;
  const float* base = emb + (size_t)t * EMB_ROWS * EMB_D;
  float4 acc = make_float4(0.f, 0.f, 0.f, 0.f);
#pragma unroll
  for (int l = 0; l < BAG_L; ++l) {
    const float4 v = *(const float4*)(base + (size_t)ip[l] * EMB_D + c * 4);
    acc.x += v.x; acc.y += v.y; acc.z += v.z; acc.w += v.w;
  }
  v4bf o;
  o[0] = (__bf16)acc.x; o[1] = (__bf16)acc.y; o[2] = (__bf16)acc.z; o[3] = (__bf16)acc.w;
  *(v4bf*)&zb[(size_t)b * ZCOLS + 64 + t * 64 + c * 4] = o;
}

// ---------- bf16 WMMA GEMM: out = relu(A[M,K] @ W[N,K]^T + bias), bf16 out ----------
// Tile: BM=128, BN=64, BK=32. 256 threads = 8 waves; wave w owns rows w*16..w*16+15
// across all 64 cols (4 WMMA 16x16 tiles, f32 accumulators).
// Global->LDS staging uses async-to-LDS with double buffering: tile k+1 streams in
// while tile k is consumed by v_wmma; one s_wait_asynccnt + one barrier per K step.
#define TM 128
#define TN 64
#define TK 32

__global__ __launch_bounds__(256)
void gemm_bf16(const __bf16* __restrict__ A, const __bf16* __restrict__ W,
               const float* __restrict__ bias, __bf16* __restrict__ out,
               int K, int ldo) {
  __shared__ __attribute__((aligned(16))) __bf16 lA[2][TM * TK];
  __shared__ __attribute__((aligned(16))) __bf16 lB[2][TN * TK];
  const int tid  = threadIdx.x;
  const int lane = tid & 31;
  const int wv   = tid >> 5;
  const int m0   = blockIdx.y * TM;
  const int n0   = blockIdx.x * TN;
  const int l15   = lane & 15;
  const int khalf = (lane >> 4) * 8;   // 16-bit frag layout: hi lanes hold K 8..15 / 24..31
  const v8f zero = {0.f, 0.f, 0.f, 0.f, 0.f, 0.f, 0.f, 0.f};
  v8f acc[4] = {zero, zero, zero, zero};

  // per-thread copy assignment: 2x16B of A (128x32), 1x16B of W (64x32)
  const int ra0 = tid >> 2;
  const int ra1 = (tid + 256) >> 2;
  const int rb  = tid >> 2;
  const int ca  = (tid & 3) * 8;
  const __bf16* gA0 = A + (size_t)(m0 + ra0) * K + ca;
  const __bf16* gA1 = A + (size_t)(m0 + ra1) * K + ca;
  const __bf16* gB  = W + (size_t)(n0 + rb ) * K + ca;
  unsigned dA0[2], dA1[2], dB[2];
#pragma unroll
  for (int p = 0; p < 2; ++p) {
    dA0[p] = lds_addr32(&lA[p][ra0 * TK + ca]);
    dA1[p] = lds_addr32(&lA[p][ra1 * TK + ca]);
    dB[p]  = lds_addr32(&lB[p][rb  * TK + ca]);
  }

  // prologue: stage tile 0
  async_copy_b128(dA0[0], gA0);
  async_copy_b128(dA1[0], gA1);
  async_copy_b128(dB[0],  gB);
  wait_async0();
  __syncthreads();

  int p = 0;
  for (int k0 = 0; k0 < K; k0 += TK) {
    if (k0 + TK < K) {               // stream next tile while computing this one
      async_copy_b128(dA0[p ^ 1], gA0 + k0 + TK);
      async_copy_b128(dA1[p ^ 1], gA1 + k0 + TK);
      async_copy_b128(dB[p ^ 1],  gB  + k0 + TK);
    }
    const __bf16* sA = lA[p];
    const __bf16* sB = lB[p];
    FragBF a;
    a.q[0] = *(const uint4*)&sA[(wv * 16 + l15) * TK + khalf];       // K = khalf..+7
    a.q[1] = *(const uint4*)&sA[(wv * 16 + l15) * TK + 16 + khalf];  // K = 16+khalf..
#pragma unroll
    for (int j = 0; j < 4; ++j) {
      FragBF bfr;
      const int n = j * 16 + l15;
      bfr.q[0] = *(const uint4*)&sB[n * TK + khalf];
      bfr.q[1] = *(const uint4*)&sB[n * TK + 16 + khalf];
      acc[j] = __builtin_amdgcn_wmma_f32_16x16x32_bf16(
          false, a.v, false, bfr.v, (short)0, acc[j], false, false);
    }
    wait_async0();     // my async fills of buffer p^1 landed in LDS
    __syncthreads();   // everyone's fills landed; everyone done reading buffer p
    p ^= 1;
  }

  // epilogue: C/D layout -> n = lane&15, m = vgpr + 8*(lane>=16)
  const int mhalf = (lane >> 4) * 8;
#pragma unroll
  for (int j = 0; j < 4; ++j) {
    const int col = n0 + j * 16 + l15;
    const float bv = bias[col];
#pragma unroll
    for (int r = 0; r < 8; ++r) {
      const int grow = m0 + wv * 16 + mhalf + r;
      float v = acc[j][r] + bv;
      v = v > 0.f ? v : 0.f;
      out[(size_t)grow * ldo + col] = (__bf16)v;
    }
  }
}

// ---------- final layer: y = sigmoid(t3[b,:256] . W3 + b3), one wave per row ----------
__global__ void final_layer(const __bf16* __restrict__ t3, const float* __restrict__ W3,
                            const float* __restrict__ b3, float* __restrict__ out) {
  int row  = blockIdx.x * 8 + (threadIdx.x >> 5);
  int lane = threadIdx.x & 31;
  const __bf16* r = t3 + (size_t)row * 256;
  float s = 0.f;
#pragma unroll
  for (int i = 0; i < 8; ++i) {
    int k = i * 32 + lane;
    s += (float)r[k] * W3[k];
  }
#pragma unroll
  for (int off = 16; off; off >>= 1) s += __shfl_xor(s, off, 32);
  if (lane == 0) out[row] = 1.f / (1.f + expf(-(s + b3[0])));
}

extern "C" void kernel_launch(void* const* d_in, const int* in_sizes, int n_in,
                              void* d_out, int out_size, void* d_ws, size_t ws_size,
                              hipStream_t stream) {
  (void)in_sizes; (void)n_in; (void)out_size; (void)ws_size;
  const float* dense  = (const float*)d_in[0];
  const float* emb    = (const float*)d_in[1];
  const float* bW[3]  = {(const float*)d_in[2],  (const float*)d_in[3],  (const float*)d_in[4]};
  const float* bB[3]  = {(const float*)d_in[5],  (const float*)d_in[6],  (const float*)d_in[7]};
  const float* tW[4]  = {(const float*)d_in[8],  (const float*)d_in[9],  (const float*)d_in[10], (const float*)d_in[11]};
  const float* tB[4]  = {(const float*)d_in[12], (const float*)d_in[13], (const float*)d_in[14], (const float*)d_in[15]};
  const int*   idx    = (const int*)d_in[16];
  // d_in[17] (offsets) unused: bags are fixed-size L=16 -> segment = b*16 by construction.
  float* outp = (float*)d_out;

  char* base = (char*)d_ws;
  size_t off = 0;
  auto alloc = [&](size_t bytes) { size_t o = off; off += (bytes + 255) & ~(size_t)255; return o; };
  __bf16* zb   = (__bf16*)(base + alloc((size_t)BATCH * ZCOLS * 2));  // concat [B,1728]
  __bf16* h1   = (__bf16*)(base + alloc((size_t)BATCH * 512 * 2));    // reused as t2
  __bf16* h2   = (__bf16*)(base + alloc((size_t)BATCH * 256 * 2));    // reused as t3
  __bf16* t1   = (__bf16*)(base + alloc((size_t)BATCH * 1024 * 2));
  __bf16* xpad = (__bf16*)(base + alloc((size_t)BATCH * 32 * 2));
  __bf16* wb0  = (__bf16*)(base + alloc((size_t)512 * 32 * 2));
  __bf16* wb1  = (__bf16*)(base + alloc((size_t)256 * 512 * 2));
  __bf16* wb2  = (__bf16*)(base + alloc((size_t)64 * 256 * 2));
  __bf16* wt0  = (__bf16*)(base + alloc((size_t)1024 * 1728 * 2));
  __bf16* wt1  = (__bf16*)(base + alloc((size_t)512 * 1024 * 2));
  __bf16* wt2  = (__bf16*)(base + alloc((size_t)256 * 512 * 2));
  __bf16* t2 = h1;
  __bf16* t3 = h2;

  auto cvt = [&](const float* in, __bf16* out, int rows, int K, int Kp) {
    int total = rows * Kp;
    pad_convert_bf16<<<(total + 255) / 256, 256, 0, stream>>>(in, out, rows, K, Kp);
  };
  // bf16 conversions (pad K=13 -> 32)
  cvt(dense, xpad, BATCH, 13, 32);
  cvt(bW[0], wb0, 512, 13, 32);
  cvt(bW[1], wb1, 256, 512, 512);
  cvt(bW[2], wb2, 64, 256, 256);
  cvt(tW[0], wt0, 1024, 1728, 1728);
  cvt(tW[1], wt1, 512, 1024, 1024);
  cvt(tW[2], wt2, 256, 512, 512);

  // embedding bag sums -> zb cols [64, 1728)
  emb_pool_kernel<<<(T_TABLES * BATCH) / 16, 256, 0, stream>>>(emb, idx, zb);

  // bottom MLP (relu): 13->512->256->64 ; last writes zb cols [0,64)
  gemm_bf16<<<dim3(512 / TN, BATCH / TM), 256, 0, stream>>>(xpad, wb0, bB[0], h1, 32, 512);
  gemm_bf16<<<dim3(256 / TN, BATCH / TM), 256, 0, stream>>>(h1, wb1, bB[1], h2, 512, 256);
  gemm_bf16<<<dim3(64 / TN, BATCH / TM), 256, 0, stream>>>(h2, wb2, bB[2], zb, 256, ZCOLS);

  // top MLP (relu): 1728->1024->512->256
  gemm_bf16<<<dim3(1024 / TN, BATCH / TM), 256, 0, stream>>>(zb, wt0, tB[0], t1, ZCOLS, 1024);
  gemm_bf16<<<dim3(512 / TN, BATCH / TM), 256, 0, stream>>>(t1, wt1, tB[1], t2, 1024, 512);
  gemm_bf16<<<dim3(256 / TN, BATCH / TM), 256, 0, stream>>>(t2, wt2, tB[2], t3, 512, 256);

  // final 256 -> 1 + sigmoid
  final_layer<<<BATCH / 8, 256, 0, stream>>>(t3, tW[3], tB[3], outp);
}
